// CrossAttention_50852412784863
// MI455X (gfx1250) — compile-verified, matrix operand order checked by reading, not run on hardware
//
#include <hip/hip_runtime.h>

// ---------------------------------------------------------------------------
// CDNA5 (gfx1250, wave32) cross-attention encoder layer.
// Linear attention reassociated:  attn = Q @ (K1^T V1 + K2^T V2)   (M shared
// by both streams), so all heavy math is dense bf16 WMMA GEMMs with f32 acc.
// GEMM uses double-buffered GLOBAL_LOAD_ASYNC_TO_LDS_B128 staging (ASYNCcnt).
// ---------------------------------------------------------------------------

typedef __attribute__((ext_vector_type(16))) __bf16 v16bf;
typedef __attribute__((ext_vector_type(8)))  __bf16 v8bf;
typedef __attribute__((ext_vector_type(8)))  float  v8f;

#define EMBED 512
#define HEADS 8
#define DH    64
#define PF    2048
#define BATCH 4
#define SEQL  2048
#define ROWS  (BATCH * SEQL)   // 8192

static __device__ __forceinline__ __bf16 f2bf(float f) {
  unsigned u = __builtin_bit_cast(unsigned, f);
  u += 0x7fffu + ((u >> 16) & 1u);            // round-to-nearest-even
  unsigned short h = (unsigned short)(u >> 16);
  return __builtin_bit_cast(__bf16, h);
}

struct BfPair { v8bf lo, hi; };
static __device__ __forceinline__ v16bf mkfrag(v8bf lo, v8bf hi) {
  BfPair p; p.lo = lo; p.hi = hi;
  return __builtin_bit_cast(v16bf, p);
}

// Per-lane async DMA: 16B global -> 16B LDS, tracked by ASYNCcnt.
// Generic pointers to __shared__ carry the wave-relative LDS offset in the
// low 32 bits, which is exactly what the instruction's VDST operand wants.
static __device__ __forceinline__ void async_b128(void* lds, const void* gsrc) {
  unsigned loff = (unsigned)(unsigned long long)lds;
  asm volatile("global_load_async_to_lds_b128 %0, %1, off"
               :: "v"(loff), "v"(gsrc) : "memory");
}
static __device__ __forceinline__ void wait_async_le4() {
  asm volatile("s_wait_asynccnt 0x4" ::: "memory");
}
static __device__ __forceinline__ void wait_async_0() {
  asm volatile("s_wait_asynccnt 0x0" ::: "memory");
}

// ---------------------------- f32 -> bf16 convert --------------------------
__global__ void cvt_bf16_kernel(const float* __restrict__ in,
                                __bf16* __restrict__ out, int n) {
  int i = blockIdx.x * blockDim.x + threadIdx.x;
  int stride = gridDim.x * blockDim.x;
  for (; i < n; i += stride) out[i] = f2bf(in[i]);
}

// ---------------------------------------------------------------------------
// Generic GEMM: C[M,N] = A[M,K](bf16) @ W[N,K](bf16)^T + bias.
// Block tile 128x128, 8 waves arranged 2x4, wave tile 64x32 (4x2 WMMA tiles).
// W is [out,in] row-major == k-contiguous per output column == the B^T layout
// the B-fragment consumes. Double-buffered async-to-LDS staging.
// ---------------------------------------------------------------------------
template <int RELU, int WRITE_F32, int WRITE_BF16>
__global__ __launch_bounds__(256)
void gemm_bf16_kernel(const __bf16* __restrict__ A, const __bf16* __restrict__ W,
                      const float* __restrict__ bias,
                      float* __restrict__ Cf, __bf16* __restrict__ Cb,
                      int M, int N, int K) {
  __shared__ __bf16 As[2][128][40];   // +8 halves pad, 16B-aligned rows
  __shared__ __bf16 Bs[2][128][40];

  const int tid   = threadIdx.x;
  const int wv    = tid >> 5;
  const int lane  = tid & 31;
  const int lrow  = lane & 15;
  const int lhalf = lane >> 4;
  const int wm    = wv >> 2;          // 0..1  (64 rows each)
  const int wn    = wv & 3;           // 0..3  (32 cols each)
  const int m0    = blockIdx.x * 128;
  const int n0    = blockIdx.y * 128;

  v8f acc[4][2] = {};

  const int arow = tid >> 1, aseg = tid & 1;   // 128 rows x (2 x 16 halves)
  const int nk = K >> 5;

  const __bf16* aptr = A + (size_t)(m0 + arow) * K + aseg * 16;
  const __bf16* bptr = W + (size_t)(n0 + arow) * K + aseg * 16;

  // stage k-tile 0 into buffer 0
  async_b128(&As[0][arow][aseg * 16],     aptr);
  async_b128(&As[0][arow][aseg * 16 + 8], aptr + 8);
  async_b128(&Bs[0][arow][aseg * 16],     bptr);
  async_b128(&Bs[0][arow][aseg * 16 + 8], bptr + 8);

  for (int kt = 0; kt < nk; ++kt) {
    const int cur = kt & 1;
    if (kt + 1 < nk) {                 // stage next tile into other buffer
      const __bf16* an = aptr + (kt + 1) * 32;
      const __bf16* bn = bptr + (kt + 1) * 32;
      async_b128(&As[cur ^ 1][arow][aseg * 16],     an);
      async_b128(&As[cur ^ 1][arow][aseg * 16 + 8], an + 8);
      async_b128(&Bs[cur ^ 1][arow][aseg * 16],     bn);
      async_b128(&Bs[cur ^ 1][arow][aseg * 16 + 8], bn + 8);
      wait_async_le4();                // current tile's 4 transfers done
    } else {
      wait_async_0();
    }
    __syncthreads();                   // all waves' staging visible

    v16bf a[4], b[2];
    for (int mi = 0; mi < 4; ++mi) {
      const int r = wm * 64 + mi * 16 + lrow;
      a[mi] = mkfrag(*(const v8bf*)&As[cur][r][lhalf * 8],
                     *(const v8bf*)&As[cur][r][16 + lhalf * 8]);
    }
    for (int ni = 0; ni < 2; ++ni) {
      const int r = wn * 32 + ni * 16 + lrow;
      b[ni] = mkfrag(*(const v8bf*)&Bs[cur][r][lhalf * 8],
                     *(const v8bf*)&Bs[cur][r][16 + lhalf * 8]);
    }
    for (int mi = 0; mi < 4; ++mi)
      for (int ni = 0; ni < 2; ++ni)
        acc[mi][ni] = __builtin_amdgcn_wmma_f32_16x16x32_bf16(
            false, a[mi], false, b[ni], (short)0, acc[mi][ni], false, false);
    __syncthreads();                   // done reading cur before re-staging it
  }

  for (int mi = 0; mi < 4; ++mi)
    for (int ni = 0; ni < 2; ++ni) {
      const int n  = n0 + wn * 32 + ni * 16 + lrow;
      const float bb = bias ? bias[n] : 0.f;
      #pragma unroll
      for (int r = 0; r < 8; ++r) {
        const int m = m0 + wm * 64 + mi * 16 + lhalf * 8 + r;
        float v = acc[mi][ni][r] + bb;
        if (RELU) v = v > 0.f ? v : 0.f;
        if (WRITE_F32)  Cf[(size_t)m * N + n] = v;
        if (WRITE_BF16) Cb[(size_t)m * N + n] = f2bf(v);
      }
    }
}

// ---------------------------------------------------------------------------
// M = K1^T V1 + K2^T V2 per (b,h): [64x64], reduced over L=2048 with WMMA.
// K/V head-columns are transposed into LDS so fragments read k-contiguous.
// Output stored transposed (Mt[n][k]) in bf16 for the next stage's B-frags.
// ---------------------------------------------------------------------------
__global__ __launch_bounds__(256)
void ktv_kernel(const __bf16* __restrict__ K1, const __bf16* __restrict__ V1,
                const __bf16* __restrict__ K2, const __bf16* __restrict__ V2,
                __bf16* __restrict__ Mt) {
  __shared__ __bf16 Kt[64][40];
  __shared__ __bf16 Vt[64][40];
  const int bh = blockIdx.x;            // b*8 + h
  const int b = bh >> 3, h = bh & 7;
  const int tid = threadIdx.x;
  const int wv = tid >> 5, lane = tid & 31, lrow = lane & 15, lhalf = lane >> 4;
  const int mt  = wv >> 1;              // 4x4 C-tiles, 2 per wave
  const int nt0 = (wv & 1) * 2;

  v8f acc[2] = {};

  const int lr = tid >> 3;              // 0..31 tile row (l)
  const int cs = tid & 7;               // 8-col segment

  for (int p = 0; p < 2; ++p) {
    const __bf16* Ks = p ? K2 : K1;
    const __bf16* Vs = p ? V2 : V1;
    for (int l0 = 0; l0 < SEQL; l0 += 32) {
      const size_t gbase = (size_t)(b * SEQL + l0 + lr) * EMBED + h * DH + cs * 8;
      v8bf kq = *(const v8bf*)(Ks + gbase);
      v8bf vq = *(const v8bf*)(Vs + gbase);
      #pragma unroll
      for (int j = 0; j < 8; ++j) {     // transpose into LDS
        Kt[cs * 8 + j][lr] = kq[j];
        Vt[cs * 8 + j][lr] = vq[j];
      }
      __syncthreads();
      const int ar = mt * 16 + lrow;
      v16bf a = mkfrag(*(const v8bf*)&Kt[ar][lhalf * 8],
                       *(const v8bf*)&Kt[ar][16 + lhalf * 8]);
      for (int ti = 0; ti < 2; ++ti) {
        const int br = (nt0 + ti) * 16 + lrow;
        v16bf bv = mkfrag(*(const v8bf*)&Vt[br][lhalf * 8],
                          *(const v8bf*)&Vt[br][16 + lhalf * 8]);
        acc[ti] = __builtin_amdgcn_wmma_f32_16x16x32_bf16(
            false, a, false, bv, (short)0, acc[ti], false, false);
      }
      __syncthreads();
    }
  }
  for (int ti = 0; ti < 2; ++ti) {
    const int col = (nt0 + ti) * 16 + lrow;       // n
    #pragma unroll
    for (int r = 0; r < 8; ++r) {
      const int row = mt * 16 + lhalf * 8 + r;    // k
      Mt[(size_t)bh * 4096 + (size_t)col * 64 + row] = f2bf(acc[ti][r]);
    }
  }
}

// ---------------------------------------------------------------------------
// attn = Q @ M per (b,h): [2048 x 64], K=64 (2 WMMA k-steps), no LDS needed.
// ---------------------------------------------------------------------------
__global__ __launch_bounds__(256)
void qm_kernel(const __bf16* __restrict__ Q, const __bf16* __restrict__ Mt,
               __bf16* __restrict__ attn) {
  const int bh = blockIdx.x;
  const int b = bh >> 3, h = bh & 7;
  const int tid = threadIdx.x;
  const int wv = tid >> 5, lane = tid & 31, lrow = lane & 15, lhalf = lane >> 4;
  const int l0 = blockIdx.y * 128 + wv * 16;

  v8f acc[4] = {};

  for (int kt = 0; kt < 2; ++kt) {
    const int k0 = kt * 32;
    const __bf16* qb = Q + (size_t)(b * SEQL + l0 + lrow) * EMBED + h * DH + k0;
    v16bf a = mkfrag(*(const v8bf*)(qb + lhalf * 8),
                     *(const v8bf*)(qb + 16 + lhalf * 8));
    for (int nt = 0; nt < 4; ++nt) {
      const __bf16* mb = Mt + (size_t)bh * 4096 + (size_t)(nt * 16 + lrow) * 64 + k0;
      v16bf bv = mkfrag(*(const v8bf*)(mb + lhalf * 8),
                        *(const v8bf*)(mb + 16 + lhalf * 8));
      acc[nt] = __builtin_amdgcn_wmma_f32_16x16x32_bf16(
          false, a, false, bv, (short)0, acc[nt], false, false);
    }
  }
  for (int nt = 0; nt < 4; ++nt) {
    #pragma unroll
    for (int r = 0; r < 8; ++r) {
      const int m = l0 + lhalf * 8 + r;
      attn[(size_t)(b * SEQL + m) * EMBED + h * DH + nt * 16 + lrow] =
          f2bf(acc[nt][r]);
    }
  }
}

// ---------------------------------------------------------------------------
// Fused residual add + LayerNorm. One wave32 per 512-wide row (8 rows/block).
// ---------------------------------------------------------------------------
__global__ __launch_bounds__(256)
void ln_kernel(const float* __restrict__ resid, const float* __restrict__ add,
               const float* __restrict__ g, const float* __restrict__ be,
               float* __restrict__ outf, int ostride, int ocol,
               __bf16* __restrict__ outb) {
  const int wv = threadIdx.x >> 5, lane = threadIdx.x & 31;
  const int row = blockIdx.x * 8 + wv;
  const float* r0 = resid + (size_t)row * EMBED;
  const float* a0 = add   + (size_t)row * EMBED;
  float v[16];
  float s = 0.f;
  #pragma unroll
  for (int i = 0; i < 16; ++i) {
    v[i] = r0[lane + 32 * i] + a0[lane + 32 * i];
    s += v[i];
  }
  #pragma unroll
  for (int m = 16; m >= 1; m >>= 1) s += __shfl_xor(s, m, 32);
  const float mu = s * (1.f / EMBED);
  float q = 0.f;
  #pragma unroll
  for (int i = 0; i < 16; ++i) { const float d = v[i] - mu; q += d * d; }
  #pragma unroll
  for (int m = 16; m >= 1; m >>= 1) q += __shfl_xor(q, m, 32);
  const float rstd = rsqrtf(q * (1.f / EMBED) + 1e-5f);
  #pragma unroll
  for (int i = 0; i < 16; ++i) {
    const int c = lane + 32 * i;
    const float y = (v[i] - mu) * rstd * g[c] + be[c];
    if (outf) outf[(size_t)row * ostride + ocol + c] = y;
    if (outb) outb[(size_t)row * EMBED + c] = f2bf(y);
  }
}

// ---------------------------------------------------------------------------
extern "C" void kernel_launch(void* const* d_in, const int* in_sizes, int n_in,
                              void* d_out, int out_size, void* d_ws, size_t ws_size,
                              hipStream_t stream) {
  (void)in_sizes; (void)n_in; (void)out_size; (void)ws_size;

  const float* Xin[2] = { (const float*)d_in[0], (const float*)d_in[1] };
  const float* Wq = (const float*)d_in[2];
  const float* bq = (const float*)d_in[3];
  const float* Wk = (const float*)d_in[4];
  const float* bk = (const float*)d_in[5];
  const float* Wv = (const float*)d_in[6];
  const float* bv = (const float*)d_in[7];
  const float* Wo = (const float*)d_in[8];
  const float* bo = (const float*)d_in[9];
  const float* ln_g = (const float*)d_in[10];
  const float* ln_b = (const float*)d_in[11];
  const float* W1 = (const float*)d_in[12];
  const float* b1 = (const float*)d_in[13];
  const float* W2 = (const float*)d_in[14];
  const float* b2 = (const float*)d_in[15];

  // ---- carve workspace -----------------------------------------------------
  char* p = (char*)d_ws;
  auto carve = [&](size_t bytes) -> void* {
    void* r = (void*)p;
    p += (bytes + 255) & ~(size_t)255;
    return r;
  };
  const size_t SEQ = (size_t)ROWS * EMBED;   // 4,194,304 elements per stream

  __bf16* bWq = (__bf16*)carve((size_t)EMBED * EMBED * 2);
  __bf16* bWk = (__bf16*)carve((size_t)EMBED * EMBED * 2);
  __bf16* bWv = (__bf16*)carve((size_t)EMBED * EMBED * 2);
  __bf16* bWo = (__bf16*)carve((size_t)EMBED * EMBED * 2);
  __bf16* bW1 = (__bf16*)carve((size_t)PF * EMBED * 2);
  __bf16* bW2 = (__bf16*)carve((size_t)EMBED * PF * 2);
  __bf16 *bX[2], *Qb[2], *Kb[2], *Vb[2], *at[2], *bx[2];
  float  *xf[2];
  for (int s = 0; s < 2; ++s) bX[s] = (__bf16*)carve(SEQ * 2);
  for (int s = 0; s < 2; ++s) { Qb[s] = (__bf16*)carve(SEQ * 2);
                                Kb[s] = (__bf16*)carve(SEQ * 2);
                                Vb[s] = (__bf16*)carve(SEQ * 2); }
  __bf16* Mt = (__bf16*)carve((size_t)32 * 4096 * 2);
  for (int s = 0; s < 2; ++s) at[s] = (__bf16*)carve(SEQ * 2);
  for (int s = 0; s < 2; ++s) xf[s] = (float*)carve(SEQ * 4);
  for (int s = 0; s < 2; ++s) bx[s] = (__bf16*)carve(SEQ * 2);
  __bf16* hb = (__bf16*)carve((size_t)ROWS * PF * 2);   // FFN mid (reused per s)
  float*  ff = (float*)carve(SEQ * 4);                  // f32 GEMM out (reused)

  // ---- bf16 conversions ----------------------------------------------------
  auto cvt = [&](const float* in, __bf16* out, size_t n) {
    cvt_bf16_kernel<<<dim3((unsigned)(n / 256)), 256, 0, stream>>>(in, out, (int)n);
  };
  cvt(Xin[0], bX[0], SEQ);
  cvt(Xin[1], bX[1], SEQ);
  cvt(Wq, bWq, (size_t)EMBED * EMBED);
  cvt(Wk, bWk, (size_t)EMBED * EMBED);
  cvt(Wv, bWv, (size_t)EMBED * EMBED);
  cvt(Wo, bWo, (size_t)EMBED * EMBED);
  cvt(W1, bW1, (size_t)PF * EMBED);
  cvt(W2, bW2, (size_t)EMBED * PF);

  // ---- QKV projections (bf16 out) -----------------------------------------
  for (int s = 0; s < 2; ++s) {
    dim3 g(ROWS / 128, EMBED / 128);
    gemm_bf16_kernel<0,0,1><<<g, 256, 0, stream>>>(bX[s], bWq, bq, nullptr, Qb[s], ROWS, EMBED, EMBED);
    gemm_bf16_kernel<0,0,1><<<g, 256, 0, stream>>>(bX[s], bWk, bk, nullptr, Kb[s], ROWS, EMBED, EMBED);
    gemm_bf16_kernel<0,0,1><<<g, 256, 0, stream>>>(bX[s], bWv, bv, nullptr, Vb[s], ROWS, EMBED, EMBED);
  }

  // ---- shared M = K1^T V1 + K2^T V2 per (b,h) ------------------------------
  ktv_kernel<<<32, 256, 0, stream>>>(Kb[0], Vb[0], Kb[1], Vb[1], Mt);

  // ---- per-stream: attn, Wo, LN, FFN, LN -> concat out ---------------------
  for (int s = 0; s < 2; ++s) {
    qm_kernel<<<dim3(32, 16), 256, 0, stream>>>(Qb[s], Mt, at[s]);

    dim3 go(ROWS / 128, EMBED / 128);
    gemm_bf16_kernel<0,1,0><<<go, 256, 0, stream>>>(at[s], bWo, bo, ff, nullptr, ROWS, EMBED, EMBED);
    ln_kernel<<<ROWS / 8, 256, 0, stream>>>(Xin[s], ff, ln_g, ln_b, xf[s], EMBED, 0, bx[s]);

    dim3 g1(ROWS / 128, PF / 128);
    gemm_bf16_kernel<1,0,1><<<g1, 256, 0, stream>>>(bx[s], bW1, b1, nullptr, hb, ROWS, PF, EMBED);
    dim3 g2(ROWS / 128, EMBED / 128);
    gemm_bf16_kernel<0,1,0><<<g2, 256, 0, stream>>>(hb, bW2, b2, ff, nullptr, ROWS, EMBED, PF);

    ln_kernel<<<ROWS / 8, 256, 0, stream>>>(xf[s], ff, ln_g, ln_b,
                                            (float*)d_out, 2 * EMBED, s * EMBED, nullptr);
  }
}
